// HashGrid_17746804867470
// MI455X (gfx1250) — matrix operand order repaired and checked by reading, not compiled, not run on hardware
//
#include <hip/hip_runtime.h>

typedef __attribute__((ext_vector_type(16))) _Float16 v16h;
typedef __attribute__((ext_vector_type(8)))  _Float16 v8h;
typedef __attribute__((ext_vector_type(2)))  _Float16 v2h;
typedef __attribute__((ext_vector_type(8)))  float    v8f;

#define NLEV   16
#define LOG2T  19
#define TSZ    (1u << LOG2T)
#define IN_CH  32
#define NW     64
#define N_PTS  1048576

// LDS strides in halves: 16-byte aligned rows, skewed to avoid bank conflicts
#define ENC_S 40
#define H_S   72
#define W0_S  40
#define W1_S  72
#define WO_S  72

// res[l] = floor(16 * (128^(1/15))^l), matches reference growth schedule
__constant__ float c_res[NLEV] = {16.f, 22.f, 30.f, 42.f, 58.f, 80.f, 111.f, 153.f,
                                  212.f, 294.f, 406.f, 561.f, 776.f, 1072.f, 1482.f, 2048.f};

__device__ __forceinline__ v16h cat8(v8h lo, v8h hi) {
  return __builtin_shufflevector(lo, hi, 0,1,2,3,4,5,6,7,8,9,10,11,12,13,14,15);
}
// A-fragment (16x32 f16, MxK): lane row = lane&15, lh = lane>>4.
// lanes 0-15 hold K 0..7 & 16..23; lanes 16-31 hold K 8..15 & 24..31.
__device__ __forceinline__ v16h lds_afrag(const _Float16* rowp, int lh) {
  v8h lo = *(const v8h*)(rowp + lh * 8);
  v8h hi = *(const v8h*)(rowp + 16 + lh * 8);
  return cat8(lo, hi);
}
// B-fragment (32x16 f16, KxN): column = lane&15; lanes 0-15 hold K 0..7 & 16..23,
// lanes 16-31 hold K 8..15 & 24..31 (same K order as the A-fragment).
__device__ __forceinline__ v16h lds_bfrag(const _Float16* colp, int lh) {
  v8h lo = *(const v8h*)(colp + lh * 8);
  v8h hi = *(const v8h*)(colp + 16 + lh * 8);
  return cat8(lo, hi);
}
// Per-lane bias lives in the same position in all 8 C/D registers (rows in
// VGPRs, columns in lanes), so bias rides in WMMA's SRC2 accumulator for free.
__device__ __forceinline__ v8f splat8(float b) {
  v8f r;
  #pragma unroll
  for (int i = 0; i < 8; ++i) r[i] = b;
  return r;
}
#define WMMA(a, b, c) \
  __builtin_amdgcn_wmma_f32_16x16x32_f16(false, (a), false, (b), (short)0, (c), false, false)

__global__ __launch_bounds__(256) void hashgrid_mlp_kernel(
    const float* __restrict__ xyzs, const float* __restrict__ tables,
    const float* __restrict__ W0, const float* __restrict__ b0,
    const float* __restrict__ W1, const float* __restrict__ b1,
    const float* __restrict__ Wo, const float* __restrict__ bo,
    float* __restrict__ out)
{
  __shared__ _Float16 sW0[NW * W0_S];     // column-major: sW0[n*W0_S + k]
  __shared__ _Float16 sW1[NW * W1_S];
  __shared__ _Float16 sWo[16 * WO_S];
  __shared__ float    sB0[NW], sB1[NW], sBo[16];
  __shared__ _Float16 sEnc[8][32 * ENC_S]; // per-wave 32x32 encode tile
  __shared__ _Float16 sH[8][16 * H_S];     // per-wave 16x64 activation tile

  const int tid  = threadIdx.x;
  const int lane = tid & 31;
  const int wv   = tid >> 5;
  const int row  = lane & 15;
  const int lh   = lane >> 4;

  // ---- stage weights: f32 global -> f16 column-major LDS ----
  for (int i = tid; i < IN_CH * NW; i += 256) { int k = i >> 6, n = i & 63; sW0[n * W0_S + k] = (_Float16)W0[i]; }
  for (int i = tid; i < NW * NW;   i += 256) { int k = i >> 6, n = i & 63; sW1[n * W1_S + k] = (_Float16)W1[i]; }
  for (int i = tid; i < NW * 16;   i += 256) { int k = i >> 4, n = i & 15; sWo[n * WO_S + k] = (_Float16)Wo[i]; }
  if (tid < NW) { sB0[tid] = b0[tid]; sB1[tid] = b1[tid]; }
  if (tid < 16) { sBo[tid] = bo[tid]; }
  __syncthreads();

  // ---- hash-grid encode: one point per lane, gathers hit L2 (table = 64MB < 192MB L2) ----
  const int g = blockIdx.x * 256 + tid;
  const float px3 = xyzs[3 * g + 0], py3 = xyzs[3 * g + 1], pz3 = xyzs[3 * g + 2];
  const float x = fminf(fmaxf((px3 + 1.f) * 0.5f, 0.f), 1.f);
  const float y = fminf(fmaxf((py3 + 1.f) * 0.5f, 0.f), 1.f);
  const float z = fminf(fmaxf((pz3 + 1.f) * 0.5f, 0.f), 1.f);
  _Float16* erow = &sEnc[wv][lane * ENC_S];
  const float2* tbl2 = (const float2*)tables;

  #pragma unroll 2   // 16 gathers in flight per lane: this loop is the roofline
  for (int l = 0; l < NLEV; ++l) {
    const float r = c_res[l];
    float px = x * r, py = y * r, pz = z * r;
    float fx = floorf(px), fy = floorf(py), fz = floorf(pz);
    float wx = px - fx, wy = py - fy, wz = pz - fz;
    unsigned cx = (unsigned)fx, cy = (unsigned)fy, cz = (unsigned)fz;
    const float2* tl = tbl2 + (size_t)l * TSZ;
    float f0 = 0.f, f1 = 0.f;
    #pragma unroll
    for (int c = 0; c < 8; ++c) {
      unsigned hx = cx + (c & 1);
      unsigned hy = cy + ((c >> 1) & 1);
      unsigned hz = cz + ((c >> 2) & 1);
      unsigned h  = (hx * 1u) ^ (hy * 2654435761u) ^ (hz * 805459861u);
      h &= (TSZ - 1u);
      float2 f = tl[h];                    // global_load_b64, L2-resident
      float w = ((c & 1)        ? wx : 1.f - wx)
              * (((c >> 1) & 1) ? wy : 1.f - wy)
              * (((c >> 2) & 1) ? wz : 1.f - wz);
      f0 = fmaf(w, f.x, f0);
      f1 = fmaf(w, f.y, f1);
    }
    v2h pk; pk[0] = (_Float16)f0; pk[1] = (_Float16)f1;
    *(v2h*)(erow + 2 * l) = pk;            // one ds_store_b32 per level
  }

  // ---- tiny MLP via v_wmma_f32_16x16x32_f16: two 16-point subtiles per wave ----
  for (int s = 0; s < 2; ++s) {
    // layer 0: enc(16x32) @ W0(32x64), K=32 in a single WMMA per N-tile
    v16h a0 = lds_afrag(&sEnc[wv][(s * 16 + row) * ENC_S], lh);
    #pragma unroll
    for (int nt = 0; nt < 4; ++nt) {
      v16h b  = lds_bfrag(&sW0[(nt * 16 + row) * W0_S], lh);
      v8f acc = WMMA(a0, b, splat8(sB0[nt * 16 + row]));   // bias in SRC2
      #pragma unroll
      for (int q = 0; q < 8; ++q) {
        float v = fmaxf(acc[q], 0.f);
        sH[wv][(q + lh * 8) * H_S + nt * 16 + row] = (_Float16)v;
      }
    }

    // layer 1: h(16x64) @ W1(64x64), K split into two 16x32 A-fragments
    v16h a1a = lds_afrag(&sH[wv][row * H_S], lh);
    v16h a1b = lds_afrag(&sH[wv][row * H_S + 32], lh);
    v8f acc1[4];
    #pragma unroll
    for (int nt = 0; nt < 4; ++nt) {
      v8f acc = WMMA(a1a, lds_bfrag(&sW1[(nt * 16 + row) * W1_S], lh),
                     splat8(sB1[nt * 16 + row]));          // bias in SRC2
      acc = WMMA(a1b, lds_bfrag(&sW1[(nt * 16 + row) * W1_S + 32], lh), acc);
      acc1[nt] = acc;
    }
    #pragma unroll
    for (int nt = 0; nt < 4; ++nt) {
      #pragma unroll
      for (int q = 0; q < 8; ++q) {
        float v = fmaxf(acc1[nt][q], 0.f);
        sH[wv][(q + lh * 8) * H_S + nt * 16 + row] = (_Float16)v;
      }
    }

    // output layer: h(16x64) @ Wo(64x16) -> one 16x16 D tile
    v16h a2a = lds_afrag(&sH[wv][row * H_S], lh);
    v16h a2b = lds_afrag(&sH[wv][row * H_S + 32], lh);
    v8f o = WMMA(a2a, lds_bfrag(&sWo[row * WO_S], lh), splat8(sBo[row]));
    o = WMMA(a2b, lds_bfrag(&sWo[row * WO_S + 32], lh), o);
    const int gbase = blockIdx.x * 256 + wv * 32 + s * 16;
    #pragma unroll
    for (int q = 0; q < 8; ++q) {
      int m = q + lh * 8;                  // D row for VGPR q per half-wave
      size_t gp = (size_t)(gbase + m);
      float v = o[q];
      if (row == 0) {
        out[gp] = fmaxf(v, 0.f);           // sigma = relu(o[:,0])
      } else {
        out[(size_t)N_PTS + gp * 15 + (size_t)(row - 1)] = v;  // geo features
      }
    }
  }
}

extern "C" void kernel_launch(void* const* d_in, const int* in_sizes, int n_in,
                              void* d_out, int out_size, void* d_ws, size_t ws_size,
                              hipStream_t stream) {
  (void)in_sizes; (void)n_in; (void)out_size; (void)d_ws; (void)ws_size;
  const float* xyzs   = (const float*)d_in[0];
  const float* tables = (const float*)d_in[1];
  const float* W0 = (const float*)d_in[2];
  const float* b0 = (const float*)d_in[3];
  const float* W1 = (const float*)d_in[4];
  const float* b1 = (const float*)d_in[5];
  const float* Wo = (const float*)d_in[6];
  const float* bo = (const float*)d_in[7];
  float* out = (float*)d_out;

  dim3 grid(N_PTS / 256), block(256);
  hipLaunchKernelGGL(hashgrid_mlp_kernel, grid, block, 0, stream,
                     xyzs, tables, W0, b0, W1, b1, Wo, bo, out);
}
